// AdaptiveAttention_22213570855331
// MI455X (gfx1250) — matrix operand order repaired and checked by reading, not compiled, not run on hardware
//
#include <hip/hip_runtime.h>
#include <math.h>

typedef __attribute__((ext_vector_type(16))) _Float16 v16h;
typedef __attribute__((ext_vector_type(8)))  _Float16 v8h;
typedef __attribute__((ext_vector_type(8)))  float    v8f;

#define SEQ   2048
#define HID   2048
#define BATCH 4

#define TM 128      // block tile M
#define TN 128      // block tile N
#define KB 32       // K-step (wmma 16x16x32 f16)
#define LP 40       // LDS row pitch in halves (80B, multiple of 16B)

#if __has_builtin(__builtin_amdgcn_global_load_async_to_lds_b128)
#define HAVE_ASYNC_LDS 1
typedef __attribute__((ext_vector_type(4))) int v4i;
#endif

static __device__ __forceinline__ void wait_async_then_barrier() {
#if defined(HAVE_ASYNC_LDS)
#if __has_builtin(__builtin_amdgcn_s_wait_asynccnt)
  __builtin_amdgcn_s_wait_asynccnt(0);
#else
  asm volatile("s_wait_asynccnt 0" ::: "memory");
#endif
#endif
  __syncthreads();
}

static __device__ __forceinline__ v16h cat16(v8h a, v8h b) {
  return __builtin_shufflevector(a, b, 0,1,2,3,4,5,6,7,8,9,10,11,12,13,14,15);
}

// Stage a 128x32 f32 tile (row-major, ld=HID) into f16 LDS [128][LP].
static __device__ __forceinline__ void stage_f32_32(_Float16 (*dst)[LP],
    const float* __restrict__ src, size_t row0, int k0, int tid) {
  #pragma unroll
  for (int i = 0; i < 4; ++i) {              // 128 rows * 8 float4 / 256 thr
    const int idx = tid + i * 256;
    const int r   = idx >> 3;                // 0..127
    const int c   = (idx & 7) << 2;          // 0,4,..,28
    const float4 v4 = *(const float4*)(src + (row0 + r) * HID + k0 + c);
    dst[r][c+0] = (_Float16)v4.x; dst[r][c+1] = (_Float16)v4.y;
    dst[r][c+2] = (_Float16)v4.z; dst[r][c+3] = (_Float16)v4.w;
  }
}

// ---------------------------------------------------------------------------
// Kernel 1: scores = (Q @ K^T) * scale * sigmoid(span[k])   (one batch slice)
// NT gemm, inner dim = HID. Block 128x128, wave 32x64, double-buffered LDS.
// ---------------------------------------------------------------------------
__global__ __launch_bounds__(256, 2)
void qk_scores_kernel(const float* __restrict__ Q, const float* __restrict__ Km,
                      const float* __restrict__ span, float* __restrict__ Sc)
{
  __shared__ __align__(16) _Float16 lA[2][TM][LP];   // Q tile  128x32 f16
  __shared__ __align__(16) _Float16 lB[2][TN][LP];   // K tile  128x32 f16

  const int tid  = threadIdx.x;
  const int lane = tid & 31;
  const int l    = lane & 15;
  const bool hi  = lane >= 16;
  const int wave = tid >> 5;
  const int m0   = (wave >> 1) * 32;   // 0,32,64,96
  const int n0   = (wave &  1) * 64;   // 0,64
  const size_t bm = (size_t)blockIdx.x * TM;
  const size_t bn = (size_t)blockIdx.y * TN;

  v8f acc[2][4] = {};

  stage_f32_32(lA[0], Q,  bm, 0, tid);
  stage_f32_32(lB[0], Km, bn, 0, tid);

  for (int k0 = 0; k0 < HID; k0 += KB) {
    const int cur = (k0 >> 5) & 1;
    __syncthreads();                    // cur staged; prev buffer's reads done
    if (k0 + KB < HID) {
      stage_f32_32(lA[cur ^ 1], Q,  bm, k0 + KB, tid);
      stage_f32_32(lB[cur ^ 1], Km, bn, k0 + KB, tid);
    }
    if (k0 + 2 * KB < HID) {            // L2 prefetch of the K+2 tiles
      __builtin_prefetch(Q  + (bm + (tid >> 1)) * HID + k0 + 2 * KB, 0, 1);
      __builtin_prefetch(Km + (bn + (tid >> 1)) * HID + k0 + 2 * KB, 0, 1);
    }

    // A 16x32 f16 layout: lanes<16 row=l K0..15; lanes>=16 K16..31
    // B 32x16 f16 layout: lanes<16 col=l K0..15; lanes>=16 K16..31
    const int offA = hi ? 16 : 0;
    v16h af[2], bf[4];
    #pragma unroll
    for (int s = 0; s < 2; ++s) {
      const _Float16* pr = &lA[cur][m0 + s * 16 + l][0];
      af[s] = cat16(*(const v8h*)(pr + offA), *(const v8h*)(pr + offA + 8));
    }
    #pragma unroll
    for (int j = 0; j < 4; ++j) {
      const _Float16* pc = &lB[cur][n0 + j * 16 + l][hi ? 16 : 0];
      bf[j] = cat16(*(const v8h*)pc, *(const v8h*)(pc + 8));
    }
    #pragma unroll
    for (int i = 0; i < 2; ++i)
      #pragma unroll
      for (int j = 0; j < 4; ++j)
        acc[i][j] = __builtin_amdgcn_wmma_f32_16x16x32_f16(
            false, af[i], false, bf[j], (short)0, acc[i][j], false, false);
  }

  const float scale = 0.022097086912079608f;   // 1/sqrt(2048)
  #pragma unroll
  for (int j = 0; j < 4; ++j) {
    const int n = (int)bn + n0 + j * 16 + l;   // key index (last axis)
    const float gate = scale * __builtin_amdgcn_rcpf(1.0f + __expf(-span[n]));
    #pragma unroll
    for (int i = 0; i < 2; ++i) {
      const size_t mbase = bm + m0 + i * 16 + (hi ? 8 : 0);
      #pragma unroll
      for (int r = 0; r < 8; ++r)
        Sc[(mbase + r) * SEQ + n] = acc[i][j][r] * gate;
    }
  }
}

// ---------------------------------------------------------------------------
// Kernel 2: row softmax, f32 scores -> f16 weights. One block (8 waves) / row.
// Each thread owns 8 contiguous elements: 2x float4 load, 1x b128 f16 store.
// ---------------------------------------------------------------------------
__global__ __launch_bounds__(256)
void softmax_kernel(const float* __restrict__ Sc, _Float16* __restrict__ W)
{
  __shared__ float redm[8];
  __shared__ float reds[8];
  const int tid = threadIdx.x;
  const size_t row = blockIdx.x;
  const float* p = Sc + row * SEQ + tid * 8;

  const float4 a = *(const float4*)(p);
  const float4 b = *(const float4*)(p + 4);
  float v[8] = {a.x, a.y, a.z, a.w, b.x, b.y, b.z, b.w};

  float m = v[0];
  #pragma unroll
  for (int i = 1; i < 8; ++i) m = fmaxf(m, v[i]);
  #pragma unroll
  for (int o = 16; o > 0; o >>= 1) m = fmaxf(m, __shfl_xor(m, o, 32));
  if ((tid & 31) == 0) redm[tid >> 5] = m;
  __syncthreads();
  m = redm[0];
  #pragma unroll
  for (int i = 1; i < 8; ++i) m = fmaxf(m, redm[i]);

  float s = 0.0f;
  #pragma unroll
  for (int i = 0; i < 8; ++i) { v[i] = __expf(v[i] - m); s += v[i]; }
  #pragma unroll
  for (int o = 16; o > 0; o >>= 1) s += __shfl_xor(s, o, 32);
  if ((tid & 31) == 0) reds[tid >> 5] = s;
  __syncthreads();
  s = 0.0f;
  #pragma unroll
  for (int i = 0; i < 8; ++i) s += reds[i];

  const float inv = __builtin_amdgcn_rcpf(s);
  v8h o;
  #pragma unroll
  for (int i = 0; i < 8; ++i) o[i] = (_Float16)(v[i] * inv);
  *(v8h*)(W + row * SEQ + tid * 8) = o;
}

// ---------------------------------------------------------------------------
// Kernel 3: O = W @ V  (NN gemm, inner dim = SEQ). W f16, V f32.
// W tile staged via GLOBAL_LOAD_ASYNC_TO_LDS_B128 when the toolchain exposes
// it (pure f16 copy, no conversion needed); V tile staged transposed through
// VGPRs (needs f32->f16 convert) so B-fragment reads are contiguous b128.
// ---------------------------------------------------------------------------
__global__ __launch_bounds__(256, 2)
void pv_kernel(const _Float16* __restrict__ Wm, const float* __restrict__ V,
               float* __restrict__ O)
{
  __shared__ __align__(16) _Float16 lA [2][TM][LP];  // W tile 128x32
  __shared__ __align__(16) _Float16 lBT[2][TN][LP];  // V tile transposed [n][k]

  const int tid  = threadIdx.x;
  const int lane = tid & 31;
  const int l    = lane & 15;
  const bool hi  = lane >= 16;
  const int wave = tid >> 5;
  const int m0   = (wave >> 1) * 32;
  const int n0   = (wave &  1) * 64;
  const size_t bm = (size_t)blockIdx.x * TM;
  const size_t bn = (size_t)blockIdx.y * TN;

  auto stageW = [&](int buf, int k0) {
    #pragma unroll
    for (int i = 0; i < 2; ++i) {              // 128 rows * 4 chunks / 256 thr
      const int idx = tid + i * 256;
      const int r   = idx >> 2;                // 0..127
      const int c   = (idx & 3) << 3;          // 0,8,16,24 (halves)
#if defined(HAVE_ASYNC_LDS)
      __builtin_amdgcn_global_load_async_to_lds_b128(
          (v4i*)(Wm + (bm + r) * SEQ + k0 + c),
          (v4i*)&lA[buf][r][c], 0, 0);
#else
      *(uint4*)&lA[buf][r][c] = *(const uint4*)(Wm + (bm + r) * SEQ + k0 + c);
#endif
    }
  };
  auto stageV = [&](int buf, int k0) {
    #pragma unroll
    for (int i = 0; i < 8; ++i) {              // 32x128 f32, pair-packed b32
      const int idx = tid + i * 256;           // 0..2047
      const int c   = idx & 127;               // 0..127 (out col)
      const int kk  = (idx >> 7) << 1;         // 0,2,..,30
      const float x0 = V[(size_t)(k0 + kk)     * HID + bn + c];
      const float x1 = V[(size_t)(k0 + kk + 1) * HID + bn + c];
      lBT[buf][c][kk]     = (_Float16)x0;
      lBT[buf][c][kk + 1] = (_Float16)x1;
    }
  };

  v8f acc[2][4] = {};
  stageW(0, 0);
  stageV(0, 0);

  for (int k0 = 0; k0 < SEQ; k0 += KB) {
    const int cur = (k0 >> 5) & 1;
    wait_async_then_barrier();          // drain ASYNCcnt, then s_barrier
    if (k0 + KB < SEQ) {
      stageW(cur ^ 1, k0 + KB);
      stageV(cur ^ 1, k0 + KB);
    }
    if (k0 + 2 * KB < SEQ) {
      __builtin_prefetch(Wm + (bm + (tid >> 1)) * SEQ + k0 + 2 * KB, 0, 1);
      __builtin_prefetch(V + (size_t)(k0 + 2 * KB + (tid >> 4)) * HID + bn + (tid & 15) * 8, 0, 1);
    }

    const int offA = hi ? 16 : 0;
    v16h af[2], bf[4];
    #pragma unroll
    for (int s = 0; s < 2; ++s) {
      const _Float16* pr = &lA[cur][m0 + s * 16 + l][0];
      af[s] = cat16(*(const v8h*)(pr + offA), *(const v8h*)(pr + offA + 8));
    }
    #pragma unroll
    for (int j = 0; j < 4; ++j) {
      const _Float16* pc = &lBT[cur][n0 + j * 16 + l][hi ? 16 : 0];
      bf[j] = cat16(*(const v8h*)pc, *(const v8h*)(pc + 8));
    }
    #pragma unroll
    for (int i = 0; i < 2; ++i)
      #pragma unroll
      for (int j = 0; j < 4; ++j)
        acc[i][j] = __builtin_amdgcn_wmma_f32_16x16x32_f16(
            false, af[i], false, bf[j], (short)0, acc[i][j], false, false);
  }

  #pragma unroll
  for (int i = 0; i < 2; ++i) {
    const size_t mbase = bm + m0 + i * 16 + (hi ? 8 : 0);
    #pragma unroll
    for (int j = 0; j < 4; ++j) {
      const size_t n = bn + n0 + j * 16 + l;
      #pragma unroll
      for (int r = 0; r < 8; ++r)
        O[(mbase + r) * HID + n] = acc[i][j][r];
    }
  }
}

// ---------------------------------------------------------------------------
extern "C" void kernel_launch(void* const* d_in, const int* in_sizes, int n_in,
                              void* d_out, int out_size, void* d_ws, size_t ws_size,
                              hipStream_t stream)
{
  const float* Q    = (const float*)d_in[0];
  const float* K    = (const float*)d_in[1];
  const float* V    = (const float*)d_in[2];
  const float* span = (const float*)d_in[3];
  float* out = (float*)d_out;

  // workspace layout (reused across batches; stream-serialized):
  //   [0, 16 MiB)  : f32 scores, SEQ x SEQ
  //   [16, 24 MiB) : f16 softmax weights, SEQ x SEQ
  float*    scores  = (float*)d_ws;
  _Float16* weights = (_Float16*)((char*)d_ws + (size_t)SEQ * SEQ * sizeof(float));

  const dim3 gridG(SEQ / TM, SEQ / TN);   // 16 x 16 blocks
  for (int b = 0; b < BATCH; ++b) {
    const size_t off = (size_t)b * SEQ * HID;
    qk_scores_kernel<<<gridG, 256, 0, stream>>>(Q + off, K + off, span, scores);
    softmax_kernel<<<SEQ, 256, 0, stream>>>(scores, weights);
    pv_kernel<<<gridG, 256, 0, stream>>>(weights, V + off, out + off);
  }
}
